// VectorQuantizerEMA_38697655337579
// MI455X (gfx1250) — matrix operand order repaired and checked by reading, not compiled, not run on hardware
//
#include <hip/hip_runtime.h>
#include <hip/hip_bf16.h>

// ---------------------------------------------------------------------------
// VQ-VAE nearest-codebook lookup for MI455X (gfx1250, wave32, WMMA).
//   z_e: (32,128,64,64) f32, embedding: (512,128) f32
//   out: (32,128,64,64) f32 = embedding[argmin_k dist(z_n, E_k)]
// argmin_k(||z||^2 - 2 z.E + ||E||^2) == argmin_k(0.5||E_k||^2 - z.E_k)
// Dot products via v_wmma_f32_16x16x32_f16 (f16 in, f32 accumulate).
// Each wave owns TWO 16-row M-tiles sharing one B fragment set:
//   - halves codebook (B) traffic per WMMA
//   - two independent accumulator chains fill WMMA data-hazard slots
// ---------------------------------------------------------------------------

typedef __attribute__((ext_vector_type(16))) _Float16 v16h;
typedef __attribute__((ext_vector_type(8)))  _Float16 v8h;
typedef __attribute__((ext_vector_type(8)))  float    v8f;

#define VQ_C    128      // embedding dim
#define VQ_K    512      // num codes
#define VQ_HW   4096     // 64*64
#define VQ_N    131072   // 32*64*64
#define TILE_N  128      // rows per workgroup (4 waves x 32 rows)
#define PZ      136      // padded LDS row stride in halves (68 dwords)

// ---------------------------------------------------------------------------
// Kernel 1: codebook -> f16 (row-major) + 0.5*||E_k||^2
// grid: 512 blocks x 128 threads (block = one code row)
// ---------------------------------------------------------------------------
__global__ void vq_prep_kernel(const float* __restrict__ emb,
                               _Float16* __restrict__ emb16,
                               float* __restrict__ esq_half) {
    const int k = blockIdx.x;
    const int c = threadIdx.x;
    float v = emb[k * VQ_C + c];
    emb16[k * VQ_C + c] = (_Float16)v;

    __shared__ float red[VQ_C];
    red[c] = v * v;
    __syncthreads();
    for (int s = VQ_C / 2; s > 0; s >>= 1) {
        if (c < s) red[c] += red[c + s];
        __syncthreads();
    }
    if (c == 0) esq_half[k] = 0.5f * red[0];
}

// ---------------------------------------------------------------------------
// Kernel 2: per-row argmin over 512 codes via WMMA.
// grid: 1024 blocks x 128 threads (4 waves). Block = 128 consecutive rows
// (never crosses a batch: 128 | 4096). Wave w owns rows [w*32, w*32+32)
// as two M-tiles with independent accumulator chains.
// ---------------------------------------------------------------------------
__global__ void __launch_bounds__(128)
vq_argmin_kernel(const float* __restrict__ z_e,
                 const _Float16* __restrict__ emb16,
                 const float* __restrict__ esq_half,
                 int* __restrict__ indices) {
    __shared__ __align__(16) _Float16 s_z[TILE_N * PZ];  // ~34.8 KB
    __shared__ float s_esq[VQ_K];                        // 2 KB

    const int tid    = threadIdx.x;
    const int n_base = blockIdx.x * TILE_N;
    const int b      = n_base >> 12;        // / 4096
    const int hw     = n_base & (VQ_HW - 1);

    // --- stage z tile to LDS as f16 [row][c], coalesced global reads ------
    const float* zb = z_e + (size_t)b * VQ_C * VQ_HW + hw;  // z_e[b, 0, hw]
    #pragma unroll
    for (int i = tid; i < TILE_N * VQ_C; i += 128) {
        const int c = i >> 7;       // / 128
        const int r = i & 127;      // consecutive lanes -> consecutive hw
        s_z[r * PZ + c] = (_Float16)zb[(size_t)c * VQ_HW + r];
    }
    for (int i = tid; i < VQ_K; i += 128) s_esq[i] = esq_half[i];
    __syncthreads();

    const int lane  = tid & 31;
    const int half  = lane >> 4;        // 0: lanes 0-15, 1: lanes 16-31
    const int lrow  = lane & 15;
    const int mbase = (tid >> 5) * 32;  // wave's 32-row slab

    // --- build A fragments once (16x32 f16 layout), reuse over all k ------
    // lane element j: K = base0 + j (j<8), K = base0 + 16 + (j-8) (j>=8)
    const int base0 = half * 8;
    v16h afrag0[4], afrag1[4];
    #pragma unroll
    for (int cc = 0; cc < 4; ++cc) {
        const _Float16* p0 = &s_z[(mbase + lrow) * PZ + cc * 32 + base0];
        const _Float16* p1 = &s_z[(mbase + 16 + lrow) * PZ + cc * 32 + base0];
        v8h lo0 = *(const v8h*)p0;
        v8h hi0 = *(const v8h*)(p0 + 16);
        v8h lo1 = *(const v8h*)p1;
        v8h hi1 = *(const v8h*)(p1 + 16);
        afrag0[cc] = __builtin_shufflevector(lo0, hi0,
            0, 1, 2, 3, 4, 5, 6, 7, 8, 9, 10, 11, 12, 13, 14, 15);
        afrag1[cc] = __builtin_shufflevector(lo1, hi1,
            0, 1, 2, 3, 4, 5, 6, 7, 8, 9, 10, 11, 12, 13, 14, 15);
    }

    float minv0[8], minv1[8];
    int   mini0[8], mini1[8];
    #pragma unroll
    for (int r = 0; r < 8; ++r) {
        minv0[r] = 3.4e38f; mini0[r] = 0;
        minv1[r] = 3.4e38f; mini1[r] = 0;
    }

    const int bco = half * 8;   // B fragment K-base within a 32-chunk
    for (int kt = 0; kt < VQ_K / 16; ++kt) {
        const int k = kt * 16 + lrow;   // this lane's code column
        // B fragment: channels bco..bco+7 and bco+16..bco+23 of code k (L2-resident)
        v16h bfrag[4];
        #pragma unroll
        for (int cc = 0; cc < 4; ++cc) {
            const _Float16* pb = emb16 + k * VQ_C + cc * 32 + bco;
            v8h blo = *(const v8h*)pb;
            v8h bhi = *(const v8h*)(pb + 16);
            bfrag[cc] = __builtin_shufflevector(blo, bhi,
                0, 1, 2, 3, 4, 5, 6, 7, 8, 9, 10, 11, 12, 13, 14, 15);
        }

        // two independent accumulator chains sharing B
        v8f acc0 = {};
        v8f acc1 = {};
        #pragma unroll
        for (int cc = 0; cc < 4; ++cc) {
            acc0 = __builtin_amdgcn_wmma_f32_16x16x32_f16(
                false, afrag0[cc], false, bfrag[cc],
                (short)0, acc0, false, false);
            acc1 = __builtin_amdgcn_wmma_f32_16x16x32_f16(
                false, afrag1[cc], false, bfrag[cc],
                (short)0, acc1, false, false);
        }

        // D layout: vgpr r, lane -> row mtile + r + 8*half, col k
        const float es = s_esq[k];
        #pragma unroll
        for (int r = 0; r < 8; ++r) {
            const float sc0 = es - acc0[r];
            const float sc1 = es - acc1[r];
            if (sc0 < minv0[r]) { minv0[r] = sc0; mini0[r] = k; }
            if (sc1 < minv1[r]) { minv1[r] = sc1; mini1[r] = k; }
        }
    }

    // --- argmin butterfly within each 16-lane group -----------------------
    #pragma unroll
    for (int r = 0; r < 8; ++r) {
        float v0 = minv0[r]; int id0 = mini0[r];
        float v1 = minv1[r]; int id1 = mini1[r];
        #pragma unroll
        for (int off = 8; off >= 1; off >>= 1) {
            const float ov0 = __shfl_xor(v0, off, 16);
            const int   oi0 = __shfl_xor(id0, off, 16);
            const float ov1 = __shfl_xor(v1, off, 16);
            const int   oi1 = __shfl_xor(id1, off, 16);
            if (ov0 < v0 || (ov0 == v0 && oi0 < id0)) { v0 = ov0; id0 = oi0; }
            if (ov1 < v1 || (ov1 == v1 && oi1 < id1)) { v1 = ov1; id1 = oi1; }
        }
        if (lrow == r) {
            indices[n_base + mbase + half * 8 + r]       = id0;
            indices[n_base + mbase + 16 + half * 8 + r]  = id1;
        }
    }
}

// ---------------------------------------------------------------------------
// Kernel 3: gather + layout transform: out[b,c,h,w] = E[idx[n], c]
// Fully coalesced writes; codebook gather hits L2 (256 KB resident).
// ---------------------------------------------------------------------------
__global__ void vq_gather_kernel(const int* __restrict__ indices,
                                 const float* __restrict__ emb,
                                 float* __restrict__ out) {
    const long t  = (long)blockIdx.x * 256 + threadIdx.x;
    const int hw = (int)(t & (VQ_HW - 1));
    const int c  = (int)((t >> 12) & (VQ_C - 1));
    const int b  = (int)(t >> 19);
    const int n  = (b << 12) + hw;
    out[t] = emb[(size_t)indices[n] * VQ_C + c];
}

// ---------------------------------------------------------------------------
extern "C" void kernel_launch(void* const* d_in, const int* in_sizes, int n_in,
                              void* d_out, int out_size, void* d_ws, size_t ws_size,
                              hipStream_t stream) {
    const float* z_e = (const float*)d_in[0];     // 32*128*64*64
    const float* emb = (const float*)d_in[1];     // 512*128
    float* out = (float*)d_out;                   // 32*128*64*64

    // workspace layout
    int*      idxbuf = (int*)d_ws;                                   // 512 KB
    _Float16* emb16  = (_Float16*)((char*)d_ws + (size_t)VQ_N * 4);  // 128 KB
    float*    esq    = (float*)((char*)d_ws + (size_t)VQ_N * 4
                                             + (size_t)VQ_K * VQ_C * 2);  // 2 KB

    vq_prep_kernel<<<VQ_K, VQ_C, 0, stream>>>(emb, emb16, esq);
    vq_argmin_kernel<<<VQ_N / TILE_N, 128, 0, stream>>>(z_e, emb16, esq, idxbuf);
    vq_gather_kernel<<<(VQ_N * VQ_C) / 256, 256, 0, stream>>>(idxbuf, emb, out);
}
